// S2PAAttention_2164663517764
// MI455X (gfx1250) — matrix operand order repaired and checked by reading, not compiled, not run on hardware
//
#include <hip/hip_runtime.h>
#include <hip/hip_bf16.h>

typedef __attribute__((ext_vector_type(16))) __bf16 v16bf;
typedef __attribute__((ext_vector_type(8)))  __bf16 v8bf;
typedef __attribute__((ext_vector_type(4)))  __bf16 v4bf;
typedef __attribute__((ext_vector_type(8)))  float  v8f;

#define B_   8
#define T_   512
#define N_   16
#define M_   8
#define DQ_  192
#define DK_  768
#define TT_  16
#define DQP_ 200            // padded LDS row (bf16): bank-conflict-free b128 loads
#define DKP_ 776
#define QSCALE 0.03608439182435161f   // 768^-0.5

// ws layout (bf16 elements), fragment-swizzled B-operands:
//   frag f = tile*KC + kc ; within frag: lane-major, 16 contiguous elements/lane
#define WS_WQ 0
#define WS_WK 36864
#define WS_WV 184320
#define WS_WO 331776
#define WS_TOT 368640

__device__ __forceinline__ __bf16 bfc(float f) { return (__bf16)f; }

// A-fragment (16-bit 16x32): lane's 16 elements = two contiguous 8-elem runs.
__device__ __forceinline__ v16bf loadA(const __bf16* row, int kBase, int half) {
    v8bf lo = *(const v8bf*)(row + kBase + 8 * half);
    v8bf hi = *(const v8bf*)(row + kBase + 16 + 8 * half);
    return __builtin_shufflevector(lo, hi, 0,1,2,3,4,5,6,7,8,9,10,11,12,13,14,15);
}

// ---------------- prep: swizzle fp32 weights -> bf16 fragment layout ----------
__global__ __launch_bounds__(256) void swizzle_weights(
    const float* __restrict__ Wq, const float* __restrict__ Wk,
    const float* __restrict__ Wv, const float* __restrict__ Wo, __bf16* __restrict__ ws)
{
    int i = blockIdx.x * 256 + threadIdx.x;
    if (i >= WS_TOT) return;
    const float* W; int base, KC, Kd, j; bool kmajor = false;
    if (i < WS_WK)      { W = Wq; base = WS_WQ; KC = 6;  Kd = DQ_; j = i - WS_WQ; }
    else if (i < WS_WV) { W = Wk; base = WS_WK; KC = 6;  Kd = DQ_; j = i - WS_WK; kmajor = true; }
    else if (i < WS_WO) { W = Wv; base = WS_WV; KC = 24; Kd = DK_; j = i - WS_WV; }
    else                { W = Wo; base = WS_WO; KC = 6;  Kd = DQ_; j = i - WS_WO; }
    int e = j & 15, l = (j >> 4) & 31, f = j >> 9;
    int tile = f / KC, kc = f % KC;
    int half = l >> 4, n = l & 15;
    int kin = kc * 32 + (e & 7) + 8 * half + 16 * (e >> 3); // B layout: K = (e&7) + 8*half + 16*(e>>3)
    float v = kmajor ? W[(size_t)kin * DK_ + tile * 16 + n]       // Wk[d][kk]
                     : W[(size_t)(tile * 16 + n) * Kd + kin];     // W[n_out][k_in]
    ws[base + j] = bfc(v);                             // dst is linear = frag-major
}

// ---------------- fused S2PA attention ---------------------------------------
__global__ __launch_bounds__(256) void s2pa_fused(
    const float* __restrict__ x,      const float* __restrict__ keys,
    const float* __restrict__ values, const int*   __restrict__ key_map,
    const int*   __restrict__ pinyin, const int*   __restrict__ pinyin_map,
    const int*   __restrict__ pron_modified,
    const float* __restrict__ table,  const __bf16* __restrict__ ws,
    float* __restrict__ out)
{
    __shared__ __bf16 s_a[TT_][DQP_];    // x block -> ctx (phase reuse)
    __shared__ __bf16 s_q[TT_][DQP_];
    __shared__ __bf16 s_pcv[TT_][DKP_];  // p -> cv (phase reuse)
    __shared__ float  s_logits[TT_][N_];
    __shared__ float  s_w[TT_][N_];
    __shared__ float  s_pw[TT_][M_];
    __shared__ int    s_pin[TT_][M_];

    const int tid  = threadIdx.x;
    const int wave = tid >> 5;
    const int lane = tid & 31;
    const int half = lane >> 4;
    const int ln16 = lane & 15;

    const int b  = blockIdx.x / (T_ / TT_);
    const int t0 = (blockIdx.x % (T_ / TT_)) * TT_;

    const size_t alignOff = (size_t)B_ * DQ_ * T_;
    const size_t pronOff  = alignOff + (size_t)B_ * N_ * T_;
    const size_t pwOff    = pronOff + (size_t)B_ * DQ_ * T_;

    const v16bf* BQ = (const v16bf*)(ws + WS_WQ);
    const v16bf* BK = (const v16bf*)(ws + WS_WK);
    const v16bf* BV = (const v16bf*)(ws + WS_WV);
    const v16bf* BO = (const v16bf*)(ws + WS_WO);

    // ---- stage x block: s_a[t][c] = x[b, c, t0+t] ----
    for (int i = tid; i < TT_ * DQ_; i += 256) {
        int c = i / TT_, t = i % TT_;
        s_a[t][c] = bfc(x[((size_t)b * DQ_ + c) * T_ + t0 + t]);
    }
    __syncthreads();

    // ---- Q = scale * (Xblk @ Wq^T) : [TT x DQ], K = DQ ----
    for (int tile = wave; tile < DQ_ / 16; tile += 8) {
        v8f acc = {};
        for (int kc = 0; kc < DQ_ / 32; ++kc) {
            v16bf a  = loadA(&s_a[ln16][0], kc * 32, half);
            v16bf bm = BQ[(tile * 6 + kc) * 32 + lane];
            acc = __builtin_amdgcn_wmma_f32_16x16x32_bf16(false, a, false, bm,
                                                          (short)0, acc, false, false);
        }
        #pragma unroll
        for (int r = 0; r < 8; ++r)
            s_q[r + 8 * half][tile * 16 + ln16] = bfc(acc[r] * QSCALE);
    }
    __syncthreads();

    // ---- P = Q @ Wk : [TT x DK], K = DQ ----
    for (int tile = wave; tile < DK_ / 16; tile += 8) {
        v8f acc = {};
        for (int kc = 0; kc < DQ_ / 32; ++kc) {
            v16bf a  = loadA(&s_q[ln16][0], kc * 32, half);
            v16bf bm = BK[(tile * 6 + kc) * 32 + lane];
            acc = __builtin_amdgcn_wmma_f32_16x16x32_bf16(false, a, false, bm,
                                                          (short)0, acc, false, false);
        }
        #pragma unroll
        for (int r = 0; r < 8; ++r)
            s_pcv[r + 8 * half][tile * 16 + ln16] = bfc(acc[r]);
    }
    __syncthreads();

    // ---- logits[t][n] = keys[b,t0+t,n,:] . p[t]  (float4 stream) ----
    const float* kbase = keys + (size_t)(b * T_ + t0) * N_ * DK_;
    for (int row = wave; row < TT_ * N_; row += 8) {
        int t = row >> 4, n = row & 15;
        const float4* kr4 = (const float4*)(kbase + (size_t)row * DK_);
        __builtin_prefetch(kbase + (size_t)(row + 8) * DK_ + lane * 16, 0, 1);
        float acc = 0.f;
        #pragma unroll
        for (int j = 0; j < DK_ / 128; ++j) {
            float4 kv = kr4[lane + 32 * j];
            int c = (lane + 32 * j) * 4;
            v4bf pv = *(const v4bf*)&s_pcv[t][c];
            acc += kv.x * (float)pv[0] + kv.y * (float)pv[1]
                 + kv.z * (float)pv[2] + kv.w * (float)pv[3];
        }
        #pragma unroll
        for (int off = 16; off; off >>= 1)
            acc += __shfl_xor(acc, off, 32);
        if (lane == 0) s_logits[t][n] = acc;
    }
    __syncthreads();

    // ---- softmax + pinyin weights (threads 0..15) ----
    if (tid < TT_) {
        const int t = tid, tg = t0 + t;
        float l[N_], mx = -3.0e38f;
        #pragma unroll
        for (int n = 0; n < N_; ++n) {
            float v = s_logits[t][n];
            if (key_map[(size_t)(b * T_ + tg) * N_ + n] == 0) v = -1e30f;
            l[n] = v; mx = fmaxf(mx, v);
        }
        float s = 0.f;
        #pragma unroll
        for (int n = 0; n < N_; ++n) { l[n] = __expf(l[n] - mx); s += l[n]; }
        const float inv = 1.f / s;
        #pragma unroll
        for (int n = 0; n < N_; ++n) s_w[t][n] = l[n] * inv;

        float pwv[M_];
        #pragma unroll
        for (int m = 0; m < M_; ++m) pwv[m] = 0.f;
        for (int n = 0; n < N_; ++n)
            pwv[pinyin_map[(size_t)(b * T_ + tg) * N_ + n] & (M_ - 1)] += s_w[t][n];
        float psum = 0.f;
        #pragma unroll
        for (int m = 0; m < M_; ++m) {
            int pid = pinyin[(size_t)(b * T_ + tg) * M_ + m];
            s_pin[t][m] = pid;
            if (pid == 0) pwv[m] = 0.f;
            psum += pwv[m];
        }
        const float pinv = 1.f / (psum + 1e-8f);
        const int pm = pron_modified[b * T_ + tg];
        #pragma unroll
        for (int m = 0; m < M_; ++m) {
            float v = (pm > 0) ? ((m == pm) ? 1.f : 0.f) : pwv[m] * pinv;
            s_pw[t][m] = v;
            out[pwOff + (size_t)(b * T_ + tg) * M_ + m] = v;
        }
    }
    __syncthreads();

    if (tid < TT_ * N_) {                         // align: [B,1,N,T]
        int t = tid >> 4, n = tid & 15;
        out[alignOff + ((size_t)b * N_ + n) * T_ + t0 + t] = s_w[t][n];
    }

    // ---- cv[t][kk] = sum_n w[t][n]*values (float4 stream, reuses s_pcv) ----
    const float* vbase = values + (size_t)(b * T_ + t0) * N_ * DK_;
    for (int id = tid; id < TT_ * (DK_ / 4); id += 256) {
        int t = id / (DK_ / 4), kk = (id % (DK_ / 4)) * 4;
        float a0 = 0.f, a1 = 0.f, a2 = 0.f, a3 = 0.f;
        #pragma unroll
        for (int n = 0; n < N_; ++n) {
            float4 vv = *(const float4*)(vbase + (size_t)(t * N_ + n) * DK_ + kk);
            float w = s_w[t][n];
            a0 += w * vv.x; a1 += w * vv.y; a2 += w * vv.z; a3 += w * vv.w;
        }
        v4bf o; o[0] = bfc(a0); o[1] = bfc(a1); o[2] = bfc(a2); o[3] = bfc(a3);
        *(v4bf*)&s_pcv[t][kk] = o;
    }
    __syncthreads();

    // ---- ctx = CV @ Wv^T : [TT x DQ], K = DK  (writes s_a) ----
    for (int tile = wave; tile < DQ_ / 16; tile += 8) {
        v8f acc = {};
        for (int kc = 0; kc < DK_ / 32; ++kc) {
            v16bf a  = loadA(&s_pcv[ln16][0], kc * 32, half);
            v16bf bm = BV[(tile * 24 + kc) * 32 + lane];
            acc = __builtin_amdgcn_wmma_f32_16x16x32_bf16(false, a, false, bm,
                                                          (short)0, acc, false, false);
        }
        #pragma unroll
        for (int r = 0; r < 8; ++r)
            s_a[r + 8 * half][tile * 16 + ln16] = bfc(acc[r]);
    }
    __syncthreads();

    // ---- O = ctx @ Wo^T -> context output [B,DQ,T] ----
    for (int tile = wave; tile < DQ_ / 16; tile += 8) {
        v8f acc = {};
        for (int kc = 0; kc < DQ_ / 32; ++kc) {
            v16bf a  = loadA(&s_a[ln16][0], kc * 32, half);
            v16bf bm = BO[(tile * 6 + kc) * 32 + lane];
            acc = __builtin_amdgcn_wmma_f32_16x16x32_bf16(false, a, false, bm,
                                                          (short)0, acc, false, false);
        }
        #pragma unroll
        for (int r = 0; r < 8; ++r)
            out[((size_t)b * DQ_ + tile * 16 + ln16) * T_ + t0 + r + 8 * half] = acc[r];
    }

    // ---- pron[t][d] = sum_m pw[t][m] * table[pin[t][m]][d] (float4 gather) ----
    for (int id = tid; id < TT_ * (DQ_ / 4); id += 256) {
        int t = id / (DQ_ / 4), d = (id % (DQ_ / 4)) * 4;
        float a0 = 0.f, a1 = 0.f, a2 = 0.f, a3 = 0.f;
        #pragma unroll
        for (int m = 0; m < M_; ++m) {
            float pw = s_pw[t][m];
            float4 tv = *(const float4*)(table + (size_t)s_pin[t][m] * DQ_ + d);
            a0 += pw * tv.x; a1 += pw * tv.y; a2 += pw * tv.z; a3 += pw * tv.w;
        }
        size_t o = pronOff + ((size_t)b * DQ_ + d) * T_ + t0 + t;
        out[o] = a0; out[o + T_] = a1; out[o + 2 * T_] = a2; out[o + 3 * T_] = a3;
    }
}

extern "C" void kernel_launch(void* const* d_in, const int* in_sizes, int n_in,
                              void* d_out, int out_size, void* d_ws, size_t ws_size,
                              hipStream_t stream) {
    (void)in_sizes; (void)n_in; (void)out_size; (void)ws_size;
    const float* x      = (const float*)d_in[0];
    const float* keys   = (const float*)d_in[1];
    const float* values = (const float*)d_in[2];
    const int*   kmap   = (const int*)d_in[3];
    const int*   pin    = (const int*)d_in[4];
    const int*   pmap   = (const int*)d_in[5];
    const int*   pmod   = (const int*)d_in[6];
    const float* Wq     = (const float*)d_in[7];
    const float* Wk     = (const float*)d_in[8];
    const float* Wv     = (const float*)d_in[9];
    const float* Wo     = (const float*)d_in[10];
    const float* table  = (const float*)d_in[11];
    __bf16* ws = (__bf16*)d_ws;            // needs 737,280 bytes
    float* out = (float*)d_out;

    swizzle_weights<<<(WS_TOT + 255) / 256, 256, 0, stream>>>(Wq, Wk, Wv, Wo, ws);
    s2pa_fused<<<B_ * (T_ / TT_), 256, 0, stream>>>(x, keys, values, kmap, pin, pmap,
                                                    pmod, table, ws, out);
}
